// Qwen3MoeSparseMoeBlock_41188736369342
// MI455X (gfx1250) — matrix operand order repaired and checked by reading, not compiled, not run on hardware
//
#include <hip/hip_runtime.h>

// ---------------- problem constants ----------------
constexpr int T   = 1024;   // tokens
constexpr int Hh  = 2048;   // hidden
constexpr int E   = 64;     // experts
constexpr int Ii  = 768;    // intermediate
constexpr int TK  = 8;      // top-k
constexpr int NSLOT = T * TK;   // 8192 (token, expert) slots

// ---------------- types ----------------
typedef __bf16 bf16_t;
typedef __attribute__((ext_vector_type(16))) __bf16 v16bf;
typedef __attribute__((ext_vector_type(4)))  __bf16 v4bf;
typedef __attribute__((ext_vector_type(8)))  float  v8f;
typedef __attribute__((ext_vector_type(4)))  float  v4f;

__device__ __forceinline__ void atomic_add_f32(float* p, float v) {
    __hip_atomic_fetch_add(p, v, __ATOMIC_RELAXED, __HIP_MEMORY_SCOPE_AGENT);
}

// fast silu: g * rcp(1 + exp(-g))  (result is rounded to bf16 anyway)
__device__ __forceinline__ float fast_silu(float g) {
    return g * __builtin_amdgcn_rcpf(1.0f + __expf(-g));
}

// ---------------- kernel 0: zero output + counters ----------------
__global__ void zero_kernel(float* __restrict__ out, int* __restrict__ counts,
                            int* __restrict__ cursor) {
    int i = blockIdx.x * 256 + threadIdx.x;
    if (i < T * Hh) out[i] = 0.0f;
    if (i < E) { counts[i] = 0; cursor[i] = 0; }
}

// ---------------- kernel 1: router (logits -> top-8 -> renorm weights) ----
__global__ void router_kernel(const float* __restrict__ X,
                              const float* __restrict__ Wr,   // [H,E]
                              int* __restrict__ topk_idx,     // [T,8]
                              float* __restrict__ topk_w,     // [T,8]
                              int* __restrict__ counts) {
    int t = blockIdx.x;
    int e = threadIdx.x;
    const float* xrow = X + (size_t)t * Hh;
    float acc = 0.0f;
#pragma unroll 4
    for (int h = 0; h < Hh; ++h)
        acc = fmaf(xrow[h], Wr[(size_t)h * E + e], acc);

    __shared__ float logits[E];
    logits[e] = acc;
    __syncthreads();

    if (e == 0) {
        float mx = logits[0];
        for (int i = 1; i < E; ++i) mx = fmaxf(mx, logits[i]);
        unsigned long long used = 0ull;
        int   sel[TK];
        float sw[TK];
        float ssum = 0.0f;
        for (int k = 0; k < TK; ++k) {
            int best = 0; float bv = -__builtin_inff();
            for (int i = 0; i < E; ++i) {
                if (!((used >> i) & 1ull) && logits[i] > bv) { bv = logits[i]; best = i; }
            }
            used |= (1ull << best);
            sel[k] = best;
            float w = __expf(logits[best] - mx);   // softmax denom cancels after renorm
            sw[k] = w; ssum += w;
        }
        float inv = 1.0f / ssum;
        for (int k = 0; k < TK; ++k) {
            topk_idx[t * TK + k] = sel[k];
            topk_w  [t * TK + k] = sw[k] * inv;
            atomicAdd(&counts[sel[k]], 1);
        }
    }
}

// ---------------- kernel 2: exclusive scan over expert counts -------------
__global__ void offsets_kernel(const int* __restrict__ counts, int* __restrict__ offsets) {
    if (threadIdx.x == 0) {
        int s = 0;
        for (int e = 0; e < E; ++e) { offsets[e] = s; s += counts[e]; }
        offsets[E] = s;
    }
}

// ---------------- kernel 3: compact per-expert token lists ----------------
__global__ void compact_kernel(const int* __restrict__ topk_idx,
                               const float* __restrict__ topk_w,
                               const int* __restrict__ offsets,
                               int* __restrict__ cursor,
                               int* __restrict__ slot_token,
                               float* __restrict__ slot_w) {
    int t = blockIdx.x * blockDim.x + threadIdx.x;
    if (t >= T) return;
    for (int k = 0; k < TK; ++k) {
        int e   = topk_idx[t * TK + k];
        int pos = atomicAdd(&cursor[e], 1);
        slot_token[offsets[e] + pos] = t;
        slot_w    [offsets[e] + pos] = topk_w[t * TK + k];
    }
}

// ---------------- WMMA tiling parameters ----------------
// block = 256 threads = 8 waves, wave grid 2(M) x 4(N)
// tile: 128 (M, slots) x 128 (N); staged K = 64 = two 32-wide WMMA k-blocks
constexpr int TILE_M = 128;
constexpr int TILE_N = 128;
constexpr int KSTEP  = 64;
constexpr int KBLK   = 4096;   // elements per 128x32 (or 32x128) k-block in LDS

// WMMA A fragment (16x32 bf16 subtile = 512 elems): lane L reads 16 bf16 at
// [subtile*512 + L*16].  For K aligned to 4, the 4 elems share a lane with
// consecutive j -> packable as one b64 store.
__device__ __forceinline__ int a_lds_index(int m /*0..127*/, int k /*0..31*/) {
    int m8 = m >> 4, mm = m & 15;
    int lane = mm + (((k >> 3) & 1) << 4);
    int j    = (k & 7) + ((k >> 4) << 3);
    return m8 * 512 + lane * 16 + j;
}
// WMMA B fragment (32x16 bf16 subtile): lane = n%16 + ((k/8)%2)*16, j = k%8 + (k/16)*8
__device__ __forceinline__ int b_lds_index(int k /*0..31*/, int n /*0..127*/) {
    int n8 = n >> 4, nn = n & 15;
    int lane = nn + (((k >> 3) & 1) << 4);
    int j    = (k & 7) + ((k >> 4) << 3);
    return n8 * 512 + lane * 16 + j;
}

// ---------------- kernel 4: gate/up GEMMs + SwiGLU -> Hact (bf16) ---------
__global__ __launch_bounds__(256)
void moe_gemm1(const float* __restrict__ X,        // [T,H] fp32
               const float* __restrict__ Wg,       // [E,H,I] fp32
               const float* __restrict__ Wu,       // [E,H,I] fp32
               const int* __restrict__ counts,
               const int* __restrict__ offsets,
               const int* __restrict__ slot_token,
               bf16_t* __restrict__ Hact) {        // [NSLOT, I] bf16
    const int e  = blockIdx.z;
    const int mt = blockIdx.y;
    const int nt = blockIdx.x;
    const int cnt = counts[e];
    if (mt * TILE_M >= cnt) return;
    const int base = offsets[e] + mt * TILE_M;

    __shared__ bf16_t ldsA [TILE_M * KSTEP];   // 16 KB (2 k-blocks)
    __shared__ bf16_t ldsBg[KSTEP * TILE_N];   // 16 KB
    __shared__ bf16_t ldsBu[KSTEP * TILE_N];   // 16 KB
    __shared__ int    ldsTok[TILE_M];

    const int tid  = threadIdx.x;
    const int lane = tid & 31;
    const int wave = tid >> 5;
    const int wm   = wave >> 2;   // 0..1
    const int wn   = wave & 3;    // 0..3

    if (tid < TILE_M) {
        int r = mt * TILE_M + tid;
        ldsTok[tid] = (r < cnt) ? slot_token[offsets[e] + r] : -1;
    }
    __syncthreads();

    v8f accg[4][2], accu[4][2];
#pragma unroll
    for (int i = 0; i < 4; ++i)
#pragma unroll
        for (int j = 0; j < 2; ++j)
#pragma unroll
            for (int q = 0; q < 8; ++q) { accg[i][j][q] = 0.0f; accu[i][j][q] = 0.0f; }

    const float* wg_base = Wg + (size_t)e * Hh * Ii + (size_t)nt * TILE_N;
    const float* wu_base = Wu + (size_t)e * Hh * Ii + (size_t)nt * TILE_N;

    for (int kk = 0; kk < Hh; kk += KSTEP) {
        __syncthreads();
        // ---- stage A: gathered token rows (b128 loads, packed b64 stores)
#pragma unroll
        for (int idx = tid; idx < (TILE_M * KSTEP) / 4; idx += 256) {   // 2048
            int m = idx >> 4;                 // 16 vec4 per 64-wide row
            int k = (idx & 15) << 2;          // 0..60, aligned 4
            int tok = ldsTok[m];
            v4f v = {0.0f, 0.0f, 0.0f, 0.0f};
            if (tok >= 0) v = *(const v4f*)&X[(size_t)tok * Hh + kk + k];
            v4bf b;
            b[0] = (bf16_t)v[0]; b[1] = (bf16_t)v[1];
            b[2] = (bf16_t)v[2]; b[3] = (bf16_t)v[3];
            *(v4bf*)&ldsA[(k >> 5) * KBLK + a_lds_index(m, k & 31)] = b;
        }
        // ---- stage B (gate & up): b128 loads coalesced along I
#pragma unroll
        for (int idx = tid; idx < (KSTEP * TILE_N) / 4; idx += 256) {   // 2048
            int k  = idx >> 5;                // 32 vec4 per 128-wide row
            int n0 = (idx & 31) << 2;
            size_t go = (size_t)(kk + k) * Ii + n0;
            v4f vg = *(const v4f*)&wg_base[go];
            v4f vu = *(const v4f*)&wu_base[go];
            int lb = (k >> 5) * KBLK;
            int kl = k & 31;
#pragma unroll
            for (int c = 0; c < 4; ++c) {
                int bi = lb + b_lds_index(kl, n0 + c);
                ldsBg[bi] = (bf16_t)vg[c];
                ldsBu[bi] = (bf16_t)vu[c];
            }
        }
        // prefetch next weight tile into cache (global_prefetch_b8)
        if (kk + KSTEP < Hh) {
            size_t po = (size_t)(kk + KSTEP + (tid & 63)) * Ii + ((tid >> 6) << 5);
            __builtin_prefetch(wg_base + po, 0, 1);
            __builtin_prefetch(wu_base + po, 0, 1);
        }
        __syncthreads();

#pragma unroll
        for (int kb = 0; kb < 2; ++kb) {
            int lb = kb * KBLK;
            v16bf afrag[4];
#pragma unroll
            for (int i = 0; i < 4; ++i)
                afrag[i] = *(const v16bf*)&ldsA[lb + (wm * 4 + i) * 512 + lane * 16];
#pragma unroll
            for (int jn = 0; jn < 2; ++jn) {
                v16bf bg = *(const v16bf*)&ldsBg[lb + (wn * 2 + jn) * 512 + lane * 16];
                v16bf bu = *(const v16bf*)&ldsBu[lb + (wn * 2 + jn) * 512 + lane * 16];
#pragma unroll
                for (int i = 0; i < 4; ++i) {
                    accg[i][jn] = __builtin_amdgcn_wmma_f32_16x16x32_bf16(
                        false, afrag[i], false, bg, (short)0, accg[i][jn], false, false);
                    accu[i][jn] = __builtin_amdgcn_wmma_f32_16x16x32_bf16(
                        false, afrag[i], false, bu, (short)0, accu[i][jn], false, false);
                }
            }
        }
    }

    // epilogue: h = silu(g) * u, store bf16
    const int mh = (lane >> 4) * 8;   // M offset within 16x16 tile
    const int nn = lane & 15;
#pragma unroll
    for (int i = 0; i < 4; ++i) {
#pragma unroll
        for (int jn = 0; jn < 2; ++jn) {
#pragma unroll
            for (int r = 0; r < 8; ++r) {
                int m = wm * 64 + i * 16 + mh + r;
                if (mt * TILE_M + m < cnt) {
                    int n = wn * 32 + jn * 16 + nn;
                    float hv = fast_silu(accg[i][jn][r]) * accu[i][jn][r];
                    Hact[(size_t)(base + m) * Ii + nt * TILE_N + n] = (bf16_t)hv;
                }
            }
        }
    }
}

// ---------------- kernel 5: down GEMM + weighted atomic scatter -----------
__global__ __launch_bounds__(256)
void moe_gemm2(const bf16_t* __restrict__ Hact,    // [NSLOT, I]
               const float* __restrict__ Wd,       // [E,I,H] fp32
               const int* __restrict__ counts,
               const int* __restrict__ offsets,
               const int* __restrict__ slot_token,
               const float* __restrict__ slot_w,
               float* __restrict__ out) {          // [T,H] fp32
    const int e  = blockIdx.z;
    const int mt = blockIdx.y;
    const int nt = blockIdx.x;
    const int cnt = counts[e];
    if (mt * TILE_M >= cnt) return;
    const int base = offsets[e] + mt * TILE_M;

    __shared__ bf16_t ldsA[TILE_M * KSTEP];    // 16 KB
    __shared__ bf16_t ldsB[KSTEP * TILE_N];    // 16 KB
    __shared__ int    ldsTok[TILE_M];
    __shared__ float  ldsW[TILE_M];

    const int tid  = threadIdx.x;
    const int lane = tid & 31;
    const int wave = tid >> 5;
    const int wm   = wave >> 2;
    const int wn   = wave & 3;

    if (tid < TILE_M) {
        int r = mt * TILE_M + tid;
        bool ok = (r < cnt);
        ldsTok[tid] = ok ? slot_token[offsets[e] + r] : -1;
        ldsW[tid]   = ok ? slot_w    [offsets[e] + r] : 0.0f;
    }
    __syncthreads();

    v8f acc[4][2];
#pragma unroll
    for (int i = 0; i < 4; ++i)
#pragma unroll
        for (int j = 0; j < 2; ++j)
#pragma unroll
            for (int q = 0; q < 8; ++q) acc[i][j][q] = 0.0f;

    const float* wd_base = Wd + (size_t)e * Ii * Hh + (size_t)nt * TILE_N;

    for (int kk = 0; kk < Ii; kk += KSTEP) {
        __syncthreads();
        // ---- stage A: Hact rows already bf16 (b64 loads, packed b64 stores)
#pragma unroll
        for (int idx = tid; idx < (TILE_M * KSTEP) / 4; idx += 256) {
            int m = idx >> 4;
            int k = (idx & 15) << 2;
            v4bf v;
            if (mt * TILE_M + m < cnt)
                v = *(const v4bf*)&Hact[(size_t)(base + m) * Ii + kk + k];
            else {
                v[0] = (bf16_t)0.0f; v[1] = (bf16_t)0.0f;
                v[2] = (bf16_t)0.0f; v[3] = (bf16_t)0.0f;
            }
            *(v4bf*)&ldsA[(k >> 5) * KBLK + a_lds_index(m, k & 31)] = v;
        }
        // ---- stage B: Wd rows, b128 loads coalesced along H
#pragma unroll
        for (int idx = tid; idx < (KSTEP * TILE_N) / 4; idx += 256) {
            int k  = idx >> 5;
            int n0 = (idx & 31) << 2;
            v4f v = *(const v4f*)&wd_base[(size_t)(kk + k) * Hh + n0];
            int lb = (k >> 5) * KBLK;
            int kl = k & 31;
#pragma unroll
            for (int c = 0; c < 4; ++c)
                ldsB[lb + b_lds_index(kl, n0 + c)] = (bf16_t)v[c];
        }
        if (kk + KSTEP < Ii) {
            size_t po = (size_t)(kk + KSTEP + (tid & 63)) * Hh + ((tid >> 6) << 5);
            __builtin_prefetch(wd_base + po, 0, 1);
        }
        __syncthreads();

#pragma unroll
        for (int kb = 0; kb < 2; ++kb) {
            int lb = kb * KBLK;
            v16bf afrag[4];
#pragma unroll
            for (int i = 0; i < 4; ++i)
                afrag[i] = *(const v16bf*)&ldsA[lb + (wm * 4 + i) * 512 + lane * 16];
#pragma unroll
            for (int jn = 0; jn < 2; ++jn) {
                v16bf b = *(const v16bf*)&ldsB[lb + (wn * 2 + jn) * 512 + lane * 16];
#pragma unroll
                for (int i = 0; i < 4; ++i)
                    acc[i][jn] = __builtin_amdgcn_wmma_f32_16x16x32_bf16(
                        false, afrag[i], false, b, (short)0, acc[i][jn], false, false);
            }
        }
    }

    // epilogue: out[tok, col] += w_slot * acc
    const int mh = (lane >> 4) * 8;
    const int nn = lane & 15;
#pragma unroll
    for (int i = 0; i < 4; ++i) {
#pragma unroll
        for (int jn = 0; jn < 2; ++jn) {
#pragma unroll
            for (int r = 0; r < 8; ++r) {
                int m = wm * 64 + i * 16 + mh + r;
                int tok = ldsTok[m];
                if (tok >= 0) {
                    int col = nt * TILE_N + wn * 32 + jn * 16 + nn;
                    atomic_add_f32(&out[(size_t)tok * Hh + col], ldsW[m] * acc[i][jn][r]);
                }
            }
        }
    }
}

// ---------------- host launch ----------------
extern "C" void kernel_launch(void* const* d_in, const int* in_sizes, int n_in,
                              void* d_out, int out_size, void* d_ws, size_t ws_size,
                              hipStream_t stream) {
    (void)in_sizes; (void)n_in; (void)out_size; (void)ws_size;
    const float* X  = (const float*)d_in[0];
    const float* Wr = (const float*)d_in[1];
    const float* Wg = (const float*)d_in[2];
    const float* Wu = (const float*)d_in[3];
    const float* Wd = (const float*)d_in[4];
    float* out = (float*)d_out;

    char* ws = (char*)d_ws;
    int*    topk_idx   = (int*)ws;     ws += (size_t)NSLOT * sizeof(int);
    float*  topk_w     = (float*)ws;   ws += (size_t)NSLOT * sizeof(float);
    int*    counts     = (int*)ws;     ws += 64 * sizeof(int);
    int*    cursor     = (int*)ws;     ws += 64 * sizeof(int);
    int*    offsets    = (int*)ws;     ws += 128 * sizeof(int);
    int*    slot_token = (int*)ws;     ws += (size_t)NSLOT * sizeof(int);
    float*  slot_w     = (float*)ws;   ws += (size_t)NSLOT * sizeof(float);
    bf16_t* Hact       = (bf16_t*)ws;  // NSLOT * Ii bf16 = 12.6 MB

    zero_kernel   <<<(T * Hh + 255) / 256, 256, 0, stream>>>(out, counts, cursor);
    router_kernel <<<T, 64, 0, stream>>>(X, Wr, topk_idx, topk_w, counts);
    offsets_kernel<<<1, 32, 0, stream>>>(counts, offsets);
    compact_kernel<<<(T + 255) / 256, 256, 0, stream>>>(topk_idx, topk_w, offsets,
                                                        cursor, slot_token, slot_w);
    moe_gemm1<<<dim3(Ii / TILE_N, T / TILE_M, E), 256, 0, stream>>>(
        X, Wg, Wu, counts, offsets, slot_token, Hact);
    moe_gemm2<<<dim3(Hh / TILE_N, T / TILE_M, E), 256, 0, stream>>>(
        Hact, Wd, counts, offsets, slot_token, slot_w, out);
}
